// HMCC_87445534146605
// MI455X (gfx1250) — matrix operand order repaired and checked by reading, not compiled, not run on hardware
//
#include <hip/hip_runtime.h>

typedef __attribute__((ext_vector_type(16))) __bf16 v16bf;
typedef __attribute__((ext_vector_type(8)))  float  v8f;

#define BSZ        8192
#define DDIM       64
#define SUB_DEPTH  6
#define SUB_LEAVES 64      // 2^6
#define SUB_NODES  63
#define SUB_COLS   126     // 63 nodes * 2 children
#define COL_TILES  8       // ceil(126/16)

// One wave (32 lanes) per block. Computes a 16-row x 126-col logits GEMM with
// bf16 WMMA (fp32 accum), pairwise softmax (+bias) in LDS, then the 6-level
// subtree product, scaled by an optional prefix probability.
__global__ __launch_bounds__(32) void hnet_subtree_kernel(
    const float* __restrict__ x,      // [8192][64]
    const float* __restrict__ W,      // [4095][2][64]
    const float* __restrict__ bias,   // [4095][2]
    const float* __restrict__ prefix, // [B][64] (or null -> 1.0)
    float*       __restrict__ out,    // K1: p6 [B][64]; K2: [B][4096]
    int level_base,                   // 0 (top half) or 6 (bottom half)
    int out_stride,                   // 64 or 4096
    int out_col_scale)                // 0 or 64  (column base = subtree * scale)
{
    __shared__ float sc[16][COL_TILES * 16 + 2];

    const int lane    = threadIdx.x;        // 0..31
    const int rowbase = blockIdx.x * 16;
    const int sub     = blockIdx.y;         // subtree index at level_base

    // ---- A fragments: 16x64 tile of x, packed in contiguous k order, same as B ----
    // lane l: row = l&15, K values kb..kb+15, kb = (l>>4)*16
    const int m_row = lane & 15;
    const int kb    = (lane >> 4) * 16;
    const float* xr = x + (rowbase + m_row) * DDIM;
    v16bf a0, a1;                           // K = 0..31 and K = 32..63
    #pragma unroll
    for (int e = 0; e < 8; ++e) {
        a0[e]     = (__bf16)xr[kb + e];
        a0[e + 8] = (__bf16)xr[kb + 8 + e];
        a1[e]     = (__bf16)xr[32 + kb + e];
        a1[e + 8] = (__bf16)xr[32 + kb + 8 + e];
    }

    // B layout: lane l supplies column n=l&15; holds 16 contiguous K values at
    // offset (l>>4)*16 within each 32-wide K fragment.
    const int kseg = (lane >> 4) * 16;

    // ---- GEMM: logits[16][126] -> LDS ----
    for (int ct = 0; ct < COL_TILES; ++ct) {
        const int j = ct * 16 + (lane & 15);   // local column (node*2 + child)
        v16bf b0, b1;
        if (j < SUB_COLS) {
            const int nl = j >> 1;                       // local node 0..62
            const int d  = 31 - __clz(nl + 1);           // depth in subtree
            const int t  = nl + 1 - (1 << d);            // index within depth
            const int gnode = (1 << (level_base + d)) - 1 + (sub << d) + t;
            const float* wrow = W + gnode * (2 * DDIM) + (j & 1) * DDIM;
            #pragma unroll
            for (int e = 0; e < 16; ++e) {
                b0[e] = (__bf16)wrow[kseg + e];
                b1[e] = (__bf16)wrow[32 + kseg + e];
            }
        } else {                                         // pad columns 126,127
            #pragma unroll
            for (int e = 0; e < 16; ++e) { b0[e] = (__bf16)0.0f; b1[e] = (__bf16)0.0f; }
        }

        v8f acc = {};
        acc = __builtin_amdgcn_wmma_f32_16x16x32_bf16(false, a0, false, b0,
                                                      (short)0, acc, false, false);
        acc = __builtin_amdgcn_wmma_f32_16x16x32_bf16(false, a1, false, b1,
                                                      (short)0, acc, false, false);

        // C layout: lane l holds col n=l&15, rows 8*(l>>4)+q for q=0..7
        const int n  = lane & 15;
        const int mb = (lane >> 4) * 8;
        #pragma unroll
        for (int q = 0; q < 8; ++q)
            sc[mb + q][ct * 16 + n] = acc[q];
    }
    __syncthreads();

    // ---- pairwise softmax (+bias) in LDS ----
    for (int idx = lane; idx < 16 * SUB_NODES; idx += 32) {
        const int row = idx / SUB_NODES;
        const int nl  = idx - row * SUB_NODES;
        const int d   = 31 - __clz(nl + 1);
        const int t   = nl + 1 - (1 << d);
        const int gnode = (1 << (level_base + d)) - 1 + (sub << d) + t;
        const float l0 = sc[row][2 * nl]     + bias[gnode * 2 + 0];
        const float l1 = sc[row][2 * nl + 1] + bias[gnode * 2 + 1];
        const float mx  = fmaxf(l0, l1);
        const float e0  = __expf(l0 - mx);
        const float e1  = __expf(l1 - mx);
        const float inv = 1.0f / (e0 + e1);
        sc[row][2 * nl]     = e0 * inv;
        sc[row][2 * nl + 1] = e1 * inv;
    }
    __syncthreads();

    // ---- 6-level tree walk: leaf probabilities ----
    for (int idx = lane; idx < 16 * SUB_LEAVES; idx += 32) {
        const int row = idx >> 6;
        const int j   = idx & 63;
        float p = prefix ? prefix[(rowbase + row) * 64 + sub] : 1.0f;
        #pragma unroll
        for (int d = 0; d < SUB_DEPTH; ++d) {
            const int t   = j >> (SUB_DEPTH - d);              // node idx at depth d
            const int k   = (j >> (SUB_DEPTH - 1 - d)) & 1;    // child taken
            const int col = 2 * ((1 << d) - 1 + t) + k;
            p *= sc[row][col];
        }
        out[(rowbase + row) * out_stride + sub * out_col_scale + j] = p;
    }
}

extern "C" void kernel_launch(void* const* d_in, const int* in_sizes, int n_in,
                              void* d_out, int out_size, void* d_ws, size_t ws_size,
                              hipStream_t stream) {
    const float* x = (const float*)d_in[0];   // [8192][64]
    const float* W = (const float*)d_in[1];   // [4095][2][64]
    const float* b = (const float*)d_in[2];   // [4095][2]
    float* out = (float*)d_out;               // [8192][4096]
    float* p6  = (float*)d_ws;                // [8192][64] prefix probs (2 MB)

    const dim3 blk(32, 1, 1);
    // Phase 1: tree levels 0..5  -> prefix probabilities at the 64 level-6 nodes
    hnet_subtree_kernel<<<dim3(BSZ / 16, 1), blk, 0, stream>>>(
        x, W, b, nullptr, p6, /*level_base=*/0, /*out_stride=*/64, /*col_scale=*/0);
    // Phase 2: tree levels 6..11 -> 64 subtrees, final posterior [B][4096]
    hnet_subtree_kernel<<<dim3(BSZ / 16, 64), blk, 0, stream>>>(
        x, W, b, p6, out, /*level_base=*/6, /*out_stride=*/4096, /*col_scale=*/64);
}
